// ExLoss_6528350290482
// MI455X (gfx1250) — compile-verified, matrix operand order checked by reading, not run on hardware
//
#include <hip/hip_runtime.h>

// Problem constants (match reference)
#define B_ROWS 4096
#define D_K    256
#define C_COLS 32000
#define NCB    (C_COLS / 128)   // 250 column blocks

typedef __attribute__((ext_vector_type(16))) _Float16 v16h;
typedef __attribute__((ext_vector_type(8)))  _Float16 v8h;
typedef __attribute__((ext_vector_type(4)))  _Float16 v4h;
typedef __attribute__((ext_vector_type(8)))  float    v8f;

typedef unsigned int u32x4 __attribute__((ext_vector_type(4)));
typedef int          i32x4 __attribute__((ext_vector_type(4)));
typedef int          i32x8 __attribute__((ext_vector_type(8)));

#define BLK_M 64
#define BLK_N 128
#define LDK   264   // padded K stride in halves (528 B -> conflict-free frag loads)

#if __has_builtin(__builtin_amdgcn_tensor_load_to_lds)
#define HAVE_TDM 1
#endif

// ---- gfx1250 async global->LDS copy (16B per lane), tracked by ASYNCcnt ----
__device__ __forceinline__ void async_copy_b128(unsigned lds_off, const void* gptr) {
  asm volatile("global_load_async_to_lds_b128 %0, %1, off"
               :: "v"(lds_off), "v"((unsigned long long)(size_t)gptr)
               : "memory");
}
__device__ __forceinline__ void wait_async0() {
  asm volatile("s_wait_asynccnt 0x0" ::: "memory");
}

// Gather one 16x32 (f16) WMMA A/B fragment for this lane from an LDS row.
// Lanes 0-15: K = k0+[0..7], k0+16+[0..7]; lanes 16-31: +8 on both chunks.
__device__ __forceinline__ v16h load_frag(const _Float16* p, int hs) {
  union { v16h v; v8h h[2]; } u;
  u.h[0] = *(const v8h*)(p + hs * 8);
  u.h[1] = *(const v8h*)(p + 16 + hs * 8);
  return u.v;
}

// f32 -> f16 pre-conversion (run once per call over inputs and V)
__global__ __launch_bounds__(256) void cvt_f32_to_f16(
    const float* __restrict__ src, _Float16* __restrict__ dst, int n4) {
  const int i = blockIdx.x * 256 + threadIdx.x;
  if (i < n4) {
    const float4 f = ((const float4*)src)[i];
    v4h h;
    h.x = (_Float16)f.x; h.y = (_Float16)f.y;
    h.z = (_Float16)f.z; h.w = (_Float16)f.w;
    ((v4h*)dst)[i] = h;
  }
}

__global__ __launch_bounds__(256) void exloss_gemm(
    const _Float16* __restrict__ Ah,   // [B_ROWS, D_K] f16
    const _Float16* __restrict__ Vh,   // [C_COLS, D_K] f16
    float* __restrict__ out,           // [B_ROWS, C_COLS] logits
    float2* __restrict__ parts) {      // [B_ROWS, NCB] (rowmax, sumexp) partials
  __shared__ _Float16 sA[BLK_M * LDK];
  __shared__ _Float16 sB[BLK_N * LDK];
  __shared__ float sPm[BLK_M * 2];
  __shared__ float sPs[BLK_M * 2];

  const int tid     = threadIdx.x;
  const int rowBase = blockIdx.y * BLK_M;
  const int colBase = blockIdx.x * BLK_N;

#ifdef HAVE_TDM
  // ---- B tile via Tensor Data Mover: one descriptor moves 128x512B with
  //      LDS padding 16B per 512B row -> 528B padded stride (bank-conflict-free)
  if (tid < 32) {   // wave 0 drives the DMA; EXEC is ignored by TDM
    const unsigned long long ga =
        (unsigned long long)(size_t)(Vh + (size_t)colBase * D_K);
    u32x4 g0;
    g0[0] = 1u;                                  // count=1, user descriptor
    g0[1] = (unsigned)(size_t)&sB[0];            // lds_addr (bytes)
    g0[2] = (unsigned)ga;                        // global_addr[31:0]
    g0[3] = (unsigned)((ga >> 32) & 0x1FFFFFFull) | (2u << 30);  // addr[56:32], type=2
    i32x8 g1;
    g1[0] = (1 << 16)        // data_size = 2 bytes
          | (1 << 20)        // pad_enable
          | (6 << 22)        // pad_interval: every 128 DWORDs (512B)
          | (3 << 25);       // pad_amount: 4 DWORDs (16B)
    g1[1] = (D_K & 0xFFFF) << 16;                          // tensor_dim0 = 256
    g1[2] = (D_K >> 16) | ((C_COLS & 0xFFFF) << 16);       // tensor_dim1 = 32000
    g1[3] = (C_COLS >> 16) | (D_K << 16);                  // tile_dim0 = 256
    g1[4] = BLK_N;                                         // tile_dim1 = 128
    g1[5] = D_K;                                           // tensor_dim0_stride = 256
    g1[6] = 0;
    g1[7] = 0;
    const i32x4 gz = {0, 0, 0, 0};
#if __clang_major__ >= 23
    const i32x8 gz8 = {0, 0, 0, 0, 0, 0, 0, 0};
    __builtin_amdgcn_tensor_load_to_lds(g0, g1, gz, gz, gz8, 0);
#else
    __builtin_amdgcn_tensor_load_to_lds(g0, g1, gz, gz, 0);
#endif
  }
#endif

  // ---- A tile via async global->LDS b128: 64 x 256 halves = 8 chunks/thread
#pragma unroll
  for (int it = 0; it < (BLK_M * D_K / 8) / 256; ++it) {
    const int i  = tid + it * 256;
    const int r  = i >> 5;          // / (D_K/8)
    const int c8 = i & 31;
    async_copy_b128((unsigned)(size_t)&sA[r * LDK + c8 * 8],
                    Ah + (size_t)(rowBase + r) * D_K + c8 * 8);
  }
#ifndef HAVE_TDM
  // fallback: B tile via async b128 (128 x 256 halves = 16 chunks/thread)
#pragma unroll
  for (int it = 0; it < (BLK_N * D_K / 8) / 256; ++it) {
    const int i  = tid + it * 256;
    const int r  = i >> 5;
    const int c8 = i & 31;
    async_copy_b128((unsigned)(size_t)&sB[r * LDK + c8 * 8],
                    Vh + (size_t)(colBase + r) * D_K + c8 * 8);
  }
#endif

#ifdef HAVE_TDM
  if (tid < 32) __builtin_amdgcn_s_wait_tensorcnt(0);
#endif
  wait_async0();
  __syncthreads();

  // ---- per-wave WMMA compute ----
  const int w    = tid >> 5;      // wave 0..7
  const int lane = tid & 31;
  const int hs   = lane >> 4;     // lane half (0/1)
  const int ln   = lane & 15;

  const int mtile = w & 3;        // 4 M-tiles of 16 rows
  const int ngrp  = w >> 2;       // 2 N-groups of 64 cols

  const _Float16* aRow  = &sA[(mtile * 16 + ln) * LDK];
  const _Float16* bBase = &sB[(ngrp * 64 + ln) * LDK];

  v8f acc[4] = {};
#pragma unroll
  for (int ks = 0; ks < D_K / 32; ++ks) {
    const int k0 = ks * 32;
    const v16h a = load_frag(aRow + k0, hs);
#pragma unroll
    for (int j = 0; j < 4; ++j) {
      const v16h b = load_frag(bBase + (size_t)j * 16 * LDK + k0, hs);
      acc[j] = __builtin_amdgcn_wmma_f32_16x16x32_f16(
          false, a, false, b, (short)0, acc[j], false, false);
    }
  }

  // ---- store logits (non-temporal: streamed out, keep L2 for operands) ----
#pragma unroll
  for (int j = 0; j < 4; ++j) {
    const int col = colBase + ngrp * 64 + j * 16 + ln;
#pragma unroll
    for (int r = 0; r < 8; ++r) {
      const int row = rowBase + mtile * 16 + 8 * hs + r;
      __builtin_nontemporal_store(acc[j][r], &out[(size_t)row * C_COLS + col]);
    }
  }

  // ---- fused partial log-sum-exp over this block's 128 columns ----
#pragma unroll
  for (int r = 0; r < 8; ++r) {
    float mx = fmaxf(fmaxf(acc[0][r], acc[1][r]), fmaxf(acc[2][r], acc[3][r]));
#pragma unroll
    for (int msk = 1; msk <= 8; msk <<= 1) mx = fmaxf(mx, __shfl_xor(mx, msk));
    float sv = __expf(acc[0][r] - mx) + __expf(acc[1][r] - mx) +
               __expf(acc[2][r] - mx) + __expf(acc[3][r] - mx);
#pragma unroll
    for (int msk = 1; msk <= 8; msk <<= 1) sv += __shfl_xor(sv, msk);
    if (ln == 0) {
      const int rl = mtile * 16 + 8 * hs + r;   // local row 0..63
      sPm[rl * 2 + ngrp] = mx;
      sPs[rl * 2 + ngrp] = sv;
    }
  }
  __syncthreads();
  if (tid < BLK_M) {
    const float m0 = sPm[tid * 2],     s0 = sPs[tid * 2];
    const float m1 = sPm[tid * 2 + 1], s1 = sPs[tid * 2 + 1];
    const float nm = fmaxf(m0, m1);
    const float s  = s0 * __expf(m0 - nm) + s1 * __expf(m1 - nm);
    parts[(size_t)(rowBase + tid) * NCB + blockIdx.x] = make_float2(nm, s);
  }
}

// One block per row: combine NCB partials + target logit -> row loss.
__global__ __launch_bounds__(256) void exloss_combine(
    const float2* __restrict__ parts, const float* __restrict__ logits,
    const int* __restrict__ targets, float* __restrict__ rowloss) {
  const int row = blockIdx.x;
  const int tid = threadIdx.x;
  float m = -3.0e38f, s = 0.0f;
  if (tid < NCB) {
    const float2 p = parts[(size_t)row * NCB + tid];
    m = p.x; s = p.y;
  }
  __shared__ float sm[256], ss[256];
  sm[tid] = m; ss[tid] = s;
  __syncthreads();
  for (int off = 128; off > 0; off >>= 1) {
    if (tid < off) {
      const float m1 = sm[tid], s1 = ss[tid];
      const float m2 = sm[tid + off], s2 = ss[tid + off];
      const float nm = fmaxf(m1, m2);
      sm[tid] = nm;
      ss[tid] = s1 * __expf(m1 - nm) + s2 * __expf(m2 - nm);
    }
    __syncthreads();
  }
  if (tid == 0) {
    const float lse = sm[0] + __logf(ss[0]);
    const float xt  = logits[(size_t)row * C_COLS + targets[row]];
    rowloss[row] = lse - xt;   // -logp[target]
  }
}

__global__ __launch_bounds__(256) void exloss_final(
    const float* __restrict__ rl, float* __restrict__ out) {
  __shared__ float ss[256];
  float s = 0.0f;
  for (int i = threadIdx.x; i < B_ROWS; i += 256) s += rl[i];
  ss[threadIdx.x] = s;
  __syncthreads();
  for (int off = 128; off > 0; off >>= 1) {
    if (threadIdx.x < off) ss[threadIdx.x] += ss[threadIdx.x + off];
    __syncthreads();
  }
  if (threadIdx.x == 0) out[0] = ss[0] / (float)B_ROWS;
}

extern "C" void kernel_launch(void* const* d_in, const int* in_sizes, int n_in,
                              void* d_out, int out_size, void* d_ws, size_t ws_size,
                              hipStream_t stream) {
  (void)in_sizes; (void)n_in; (void)out_size; (void)ws_size;
  const float* inputs  = (const float*)d_in[0];   // [4096, 256]
  const int*   targets = (const int*)  d_in[1];   // [4096]
  // d_in[2..4] unused: W_MS == 0
  const float* V       = (const float*)d_in[5];   // [32000, 256]

  float* out    = (float*)d_out;      // out[0] = loss; out+1 = logits
  float* logits = out + 1;

  // d_ws layout: Ah (2 MB) | Vh (16 MB) | parts (8 MB) | rowloss (16 KB)
  char* ws = (char*)d_ws;
  _Float16* Ah      = (_Float16*)ws;
  _Float16* Vh      = (_Float16*)(ws + 2097152);
  float2*   parts   = (float2*)  (ws + 2097152 + 16384000);
  float*    rowloss = (float*)   (ws + 2097152 + 16384000 + 8192000);

  cvt_f32_to_f16<<<(B_ROWS * D_K / 4) / 256, 256, 0, stream>>>(inputs, Ah, B_ROWS * D_K / 4);
  cvt_f32_to_f16<<<(C_COLS * D_K / 4) / 256, 256, 0, stream>>>(V, Vh, C_COLS * D_K / 4);

  dim3 grid1(NCB, B_ROWS / BLK_M);   // 250 x 64 blocks
  exloss_gemm<<<grid1, 256, 0, stream>>>(Ah, Vh, logits, parts);
  exloss_combine<<<B_ROWS, 256, 0, stream>>>(parts, logits, targets, rowloss);
  exloss_final<<<1, 256, 0, stream>>>(rowloss, out);
}
